// MultiHeadAttention_5549097747196
// MI455X (gfx1250) — matrix operand order repaired and checked by reading, not compiled, not run on hardware
//
#include <hip/hip_runtime.h>
#include <hip/hip_bf16.h>

typedef __attribute__((ext_vector_type(16))) __bf16 v16bf;
typedef __attribute__((ext_vector_type(8)))  __bf16 v8bf;
typedef __attribute__((ext_vector_type(8)))  float  v8f;
typedef __attribute__((ext_vector_type(4)))  unsigned v4u;
typedef __attribute__((ext_vector_type(8)))  int      v8i_;
typedef __attribute__((ext_vector_type(4)))  int      v4i_;

#define WMMA_BF16(a, b, c) \
  __builtin_amdgcn_wmma_f32_16x16x32_bf16(false, (a), false, (b), (short)0, (c), false, false)

__device__ static inline v16bf cat8(v8bf lo, v8bf hi) {
  v16bf r;
#pragma unroll
  for (int i = 0; i < 8; i++) { r[i] = lo[i]; r[i + 8] = hi[i]; }
  return r;
}

__device__ static inline v8f zero8() {
  v8f z = {0.f, 0.f, 0.f, 0.f, 0.f, 0.f, 0.f, 0.f};
  return z;
}

// low 32 bits of a flat pointer into __shared__ = wave-logical LDS byte address
__device__ static inline unsigned lds_off(const void* p) {
  return (unsigned)(unsigned long long)p;
}

// CDNA5 async copy global -> LDS, 16B per lane, tracked by ASYNCcnt
__device__ static inline void async_b128(unsigned lds_addr, const void* g) {
  asm volatile("global_load_async_to_lds_b128 %0, %1, off"
               :: "v"(lds_addr), "v"((unsigned long long)g)
               : "memory");
}
__device__ static inline void wait_async0() {
  asm volatile("s_wait_asynccnt 0x0" ::: "memory");
}

// ---------------------------------------------------------------------------
// Tensor Data Mover: 2D bf16 tile (global -> LDS) with HW padding.
// Row payload must be 128B (64 bf16); LDS pitch becomes 144B (= Ks[64][72]).
// D# layout per CDNA5 ISA.
// This toolchain's builtin takes 6 args (g0, g1, g2, g3, extra, cpol).
// ---------------------------------------------------------------------------
__device__ static inline void tdm_load_2d_bf16(unsigned lds_byte_addr,
                                               const void* gaddr,
                                               unsigned row_elems,     // tile/tensor dim0
                                               unsigned rows,          // tile/tensor dim1
                                               unsigned row_stride_el) // dim0 stride (elems)
{
  const unsigned long long ga = (unsigned long long)gaddr;
  v4u g0;
  g0[0] = 1u;                                              // count=1, user mode
  g0[1] = lds_byte_addr;                                   // lds_addr
  g0[2] = (unsigned)(ga & 0xffffffffu);                    // global_addr[31:0]
  g0[3] = (unsigned)((ga >> 32) & 0x1ffffffu) | (2u << 30); // addr[56:32] | type=2
  v8i_ g1;
  // data_size=2B(code1), pad_enable, pad_interval=4 (32 DW = 128B), pad_amount=3 (4 DW = 16B)
  g1[0] = (int)((1u << 16) | (1u << 20) | (4u << 22) | (3u << 25));
  g1[1] = (int)((row_elems & 0xffffu) << 16);              // tensor_dim0 lo16 (barrier addr=0)
  g1[2] = (int)(((row_elems >> 16) & 0xffffu) | ((rows & 0xffffu) << 16)); // d0 hi | d1 lo
  g1[3] = (int)(((rows >> 16) & 0xffffu) | ((row_elems & 0xffffu) << 16)); // d1 hi | tile_dim0
  g1[4] = (int)(rows & 0xffffu);                           // tile_dim1 (tile_dim2=0)
  g1[5] = (int)row_stride_el;                              // tensor_dim0_stride lo32
  g1[6] = 0;                                               // stride hi16 | dim1_stride lo16
  g1[7] = 0;
  const v4i_ z4 = {0, 0, 0, 0};                            // groups 2/3 unused (2D)
  const v8i_ z8 = {0, 0, 0, 0, 0, 0, 0, 0};
  __builtin_amdgcn_tensor_load_to_lds(g0, g1, z4, z4, z8, 0);
}

// ---------------------------------------------------------------------------
// Kernel 0: one-shot f32 -> bf16 conversion of the 4 weight matrices.
// grid = (128, 4), block = 256; each thread converts 8 elements.
// ---------------------------------------------------------------------------
__global__ __launch_bounds__(256) void cvt_weights_kernel(
    const float* __restrict__ w0, const float* __restrict__ w1,
    const float* __restrict__ w2, const float* __restrict__ w3,
    __bf16* __restrict__ d0, __bf16* __restrict__ d1,
    __bf16* __restrict__ d2, __bf16* __restrict__ d3) {
  const int m = blockIdx.y;
  const float* src = (m == 0) ? w0 : (m == 1) ? w1 : (m == 2) ? w2 : w3;
  __bf16* dst = (m == 0) ? d0 : (m == 1) ? d1 : (m == 2) ? d2 : d3;
  const size_t i = ((size_t)blockIdx.x * 256 + threadIdx.x) * 8;  // < 262144
  float4 a = *(const float4*)(src + i);
  float4 b = *(const float4*)(src + i + 4);
  __bf16* d = dst + i;
  d[0] = (__bf16)a.x; d[1] = (__bf16)a.y; d[2] = (__bf16)a.z; d[3] = (__bf16)a.w;
  d[4] = (__bf16)b.x; d[5] = (__bf16)b.y; d[6] = (__bf16)b.z; d[7] = (__bf16)b.w;
}

// ---------------------------------------------------------------------------
// Kernel 1: fused QKV projection.  grid = (64, 8, 3), block = 256 (8 waves)
// Y = X[8192,512] @ W^T + b ; head-split; Q scaled by 1/8; V transposed.
// B tile staged with async-to-LDS (bf16 weights); A tile converted in-flight.
// ---------------------------------------------------------------------------
__global__ __launch_bounds__(256) void qkv_gemm_kernel(
    const float* __restrict__ xq, const float* __restrict__ xk, const float* __restrict__ xv,
    const __bf16* __restrict__ wq, const __bf16* __restrict__ wk, const __bf16* __restrict__ wv,
    const float* __restrict__ bq, const float* __restrict__ bk, const float* __restrict__ bv,
    __bf16* __restrict__ qh, __bf16* __restrict__ kh, __bf16* __restrict__ vt) {
  const int which = blockIdx.z;
  const float* X = (which == 0) ? xq : (which == 1) ? xk : xv;
  const __bf16* W = (which == 0) ? wq : (which == 1) ? wk : wv;
  const float* bias = (which == 0) ? bq : (which == 1) ? bk : bv;
  __bf16* outqk = (which == 0) ? qh : kh;

  __shared__ __align__(16) __bf16 As[128][40];  // 80B pitch (16B-aligned rows)
  __shared__ __align__(16) __bf16 Bs[64][40];

  const int tid = threadIdx.x;
  const int wave = tid >> 5, lane = tid & 31;
  const int half = lane >> 4, l16 = lane & 15;
  const int mw = wave & 3, nw = wave >> 2;  // 4x2 wave grid, each 32x32
  const int m_base = blockIdx.x * 128;
  const int n_base = blockIdx.y * 64;

  v8f acc[2][2];
#pragma unroll
  for (int i = 0; i < 2; i++)
#pragma unroll
    for (int j = 0; j < 2; j++) acc[i][j] = zero8();

  for (int k0 = 0; k0 < 512; k0 += 32) {
    __syncthreads();
    // B tile 64x32 bf16: 256 x 16B async chunks, 1 per thread
    {
      int r = tid >> 2, c = (tid & 3) * 8;
      async_b128(lds_off(&Bs[r][c]), W + (size_t)(n_base + r) * 512 + k0 + c);
    }
    // A tile 128x32 f32 -> bf16 LDS : 1024 float4, 4 per thread
    const int pf = (k0 + 32 < 512) ? 32 : 0;  // branchless prefetch offset
#pragma unroll
    for (int i = 0; i < 4; i++) {
      int idx = tid + i * 256;
      int r = idx >> 3, c4 = idx & 7;
      const float* ap = X + (size_t)(m_base + r) * 512 + k0 + c4 * 4;
      float4 f = *(const float4*)ap;
      __builtin_prefetch(ap + pf, 0, 3);
      __bf16* dst = &As[r][c4 * 4];
      dst[0] = (__bf16)f.x; dst[1] = (__bf16)f.y; dst[2] = (__bf16)f.z; dst[3] = (__bf16)f.w;
    }
    wait_async0();
    __syncthreads();

    v16bf a[2], b[2];
#pragma unroll
    for (int mt = 0; mt < 2; mt++) {
      const __bf16* row = &As[mw * 32 + mt * 16 + l16][0];
      a[mt] = cat8(*(const v8bf*)(row + half * 8), *(const v8bf*)(row + 16 + half * 8));
    }
#pragma unroll
    for (int nt = 0; nt < 2; nt++) {
      const __bf16* row = &Bs[nw * 32 + nt * 16 + l16][0];
      b[nt] = cat8(*(const v8bf*)(row + half * 8), *(const v8bf*)(row + 16 + half * 8));
    }
#pragma unroll
    for (int mt = 0; mt < 2; mt++)
#pragma unroll
      for (int nt = 0; nt < 2; nt++) acc[mt][nt] = WMMA_BF16(a[mt], b[nt], acc[mt][nt]);
  }

  const float qscale = (which == 0) ? 0.125f : 1.0f;  // fold 1/sqrt(64) into Q
#pragma unroll
  for (int mt = 0; mt < 2; mt++) {
#pragma unroll
    for (int nt = 0; nt < 2; nt++) {
      const int c = n_base + nw * 32 + nt * 16 + l16;  // output column 0..511
      const int h = c >> 6, d = c & 63;
      const float bv_ = bias[c];
#pragma unroll
      for (int r = 0; r < 8; r++) {
        const int m = m_base + mw * 32 + mt * 16 + r + half * 8;
        const int bb = m >> 12, s = m & 4095;
        const float val = (acc[mt][nt][r] + bv_) * qscale;
        if (which < 2)
          outqk[((size_t)((bb * 8 + h) * 4096 + s)) * 64 + d] = (__bf16)val;
        else
          vt[((size_t)((bb * 8 + h) * 64 + d)) * 4096 + s] = (__bf16)val;
      }
    }
  }
}

// ---------------------------------------------------------------------------
// Kernel 2: flash attention.  grid = (32, 16), block = 256 (8 waves)
// Double-buffered K/V tiles staged by the Tensor Data Mover (TENSORcnt),
// with HW LDS padding; WMMA overlaps the next tile's DMA.
// ---------------------------------------------------------------------------
__global__ __launch_bounds__(256) void attn_kernel(
    const __bf16* __restrict__ qh, const __bf16* __restrict__ kh,
    const __bf16* __restrict__ vt, __bf16* __restrict__ ao) {
  const int bh = blockIdx.y;
  const int b = bh >> 3, h = bh & 7;
  const int q0 = blockIdx.x * 128;
  const int tid = threadIdx.x;
  const int wave = tid >> 5, lane = tid & 31;
  const int half = lane >> 4, l16 = lane & 15;

  __shared__ __align__(16) __bf16 Ks[2][64][72];     // [buf][key][depth], TDM-padded
  __shared__ __align__(16) __bf16 Vs[2][64][72];     // [buf][depth][key], TDM-padded
  __shared__ __align__(16) __bf16 Ps[8][16][72];     // per-wave P

  const __bf16* qbase = qh + (size_t)bh * 4096 * 64;
  const __bf16* kbase = kh + (size_t)bh * 4096 * 64;
  const __bf16* vbase = vt + (size_t)bh * 64 * 4096;

  // one wave drives the TDM; TENSORcnt is per-wave, others wait at the barrier
  auto stage_kv = [&](int s0, int buf) {
    if (wave == 0) {
      tdm_load_2d_bf16(lds_off(&Ks[buf][0][0]), kbase + (size_t)s0 * 64, 64, 64, 64);
      tdm_load_2d_bf16(lds_off(&Vs[buf][0][0]), vbase + s0, 64, 64, 4096);
    }
  };

  // Q A-operands: rows q0 + wave*16 + l16, depth chunks per ISA A layout.
  v16bf qa[2];
  {
    const __bf16* qrow = qbase + (size_t)(q0 + wave * 16 + l16) * 64;
#pragma unroll
    for (int kk = 0; kk < 2; kk++)
      qa[kk] = cat8(*(const v8bf*)(qrow + kk * 32 + half * 8),
                    *(const v8bf*)(qrow + kk * 32 + 16 + half * 8));
  }

  v8f o[4];
  float m_run[8], l_run[8];
#pragma unroll
  for (int t = 0; t < 4; t++) o[t] = zero8();
#pragma unroll
  for (int r = 0; r < 8; r++) { m_run[r] = -1.0e30f; l_run[r] = 0.f; }

  stage_kv(0, 0);
  for (int s0 = 0; s0 < 4096; s0 += 64) {
    const int buf = (s0 >> 6) & 1;
    __builtin_amdgcn_s_wait_tensorcnt(0);  // no-op for waves with TENSORcnt==0
    __syncthreads();                       // tile visible to all waves
    if (s0 + 64 < 4096) stage_kv(s0 + 64, buf ^ 1);  // DMA next tile during compute

    // logits: 16 q rows x 64 keys = 4 tiles, depth=64 = 2 chained WMMAs
    v8f acc[4];
#pragma unroll
    for (int t = 0; t < 4; t++) {
      const __bf16* krow = &Ks[buf][t * 16 + l16][0];
      v16bf b0 = cat8(*(const v8bf*)(krow + half * 8), *(const v8bf*)(krow + 16 + half * 8));
      v16bf b1 = cat8(*(const v8bf*)(krow + 32 + half * 8), *(const v8bf*)(krow + 48 + half * 8));
      v8f z = zero8();
      z = WMMA_BF16(qa[0], b0, z);
      z = WMMA_BF16(qa[1], b1, z);
      acc[t] = z;
    }

    // online softmax: each q row lives in 16 lanes of this wave-half
    float pv[4][8], corr[8];
#pragma unroll
    for (int r = 0; r < 8; r++) {
      float mx = fmaxf(fmaxf(acc[0][r], acc[1][r]), fmaxf(acc[2][r], acc[3][r]));
#pragma unroll
      for (int off = 8; off >= 1; off >>= 1) mx = fmaxf(mx, __shfl_xor(mx, off, 32));
      const float mnew = fmaxf(m_run[r], mx);
      corr[r] = __builtin_exp2f((m_run[r] - mnew) * 1.44269504f);
      float ps = 0.f;
#pragma unroll
      for (int t = 0; t < 4; t++) {
        float p = __builtin_exp2f((acc[t][r] - mnew) * 1.44269504f);
        pv[t][r] = p; ps += p;
      }
#pragma unroll
      for (int off = 8; off >= 1; off >>= 1) ps += __shfl_xor(ps, off, 32);
      l_run[r] = l_run[r] * corr[r] + ps;
      m_run[r] = mnew;
    }

    // P (D-layout) -> per-wave LDS -> A-layout ; rescale running output
#pragma unroll
    for (int t = 0; t < 4; t++)
#pragma unroll
      for (int r = 0; r < 8; r++) {
        Ps[wave][r + half * 8][t * 16 + l16] = (__bf16)pv[t][r];
        o[t][r] *= corr[r];
      }

    v16bf pa[2];
    {
      const __bf16* prow = &Ps[wave][l16][0];
#pragma unroll
      for (int kk = 0; kk < 2; kk++)
        pa[kk] = cat8(*(const v8bf*)(prow + kk * 32 + half * 8),
                      *(const v8bf*)(prow + kk * 32 + 16 + half * 8));
    }
#pragma unroll
    for (int t = 0; t < 4; t++) {
      const __bf16* vrow = &Vs[buf][t * 16 + l16][0];  // depth row, key-contiguous
      v16bf b0 = cat8(*(const v8bf*)(vrow + half * 8), *(const v8bf*)(vrow + 16 + half * 8));
      v16bf b1 = cat8(*(const v8bf*)(vrow + 32 + half * 8), *(const v8bf*)(vrow + 48 + half * 8));
      o[t] = WMMA_BF16(pa[0], b0, o[t]);
      o[t] = WMMA_BF16(pa[1], b1, o[t]);
    }
  }

  // epilogue: normalize, write [B,S,H*64] bf16
  const int qrow = q0 + wave * 16;
#pragma unroll
  for (int r = 0; r < 8; r++) {
    const float rcp = 1.0f / l_run[r];
    const int row = qrow + r + half * 8;
#pragma unroll
    for (int t = 0; t < 4; t++)
      ao[((size_t)(b * 4096 + row)) * 512 + h * 64 + t * 16 + l16] = (__bf16)(o[t][r] * rcp);
  }
}

// ---------------------------------------------------------------------------
// Kernel 3: output projection.  grid = (64, 8), block = 256
// out = AO[8192,512](bf16) @ Wo^T + bo  (f32 out). Fully async + double-buffered.
// ---------------------------------------------------------------------------
__global__ __launch_bounds__(256) void out_proj_kernel(
    const __bf16* __restrict__ ao, const __bf16* __restrict__ wo,
    const float* __restrict__ bo, float* __restrict__ out) {
  __shared__ __align__(16) __bf16 As[2][128][40];
  __shared__ __align__(16) __bf16 Bs[2][64][40];

  const int tid = threadIdx.x;
  const int wave = tid >> 5, lane = tid & 31;
  const int half = lane >> 4, l16 = lane & 15;
  const int mw = wave & 3, nw = wave >> 2;
  const int m_base = blockIdx.x * 128;
  const int n_base = blockIdx.y * 64;

  auto stage_ab = [&](int k0, int buf) {
#pragma unroll
    for (int i = 0; i < 2; i++) {
      int idx = tid + i * 256;                 // A: 512 chunks of 8 bf16
      int r = idx >> 2, c = (idx & 3) * 8;
      async_b128(lds_off(&As[buf][r][c]), ao + (size_t)(m_base + r) * 512 + k0 + c);
    }
    int r = tid >> 2, c = (tid & 3) * 8;       // B: 256 chunks
    async_b128(lds_off(&Bs[buf][r][c]), wo + (size_t)(n_base + r) * 512 + k0 + c);
  };

  v8f acc[2][2];
#pragma unroll
  for (int i = 0; i < 2; i++)
#pragma unroll
    for (int j = 0; j < 2; j++) acc[i][j] = zero8();

  stage_ab(0, 0);
  for (int k0 = 0; k0 < 512; k0 += 32) {
    const int buf = (k0 >> 5) & 1;
    wait_async0();
    __syncthreads();
    if (k0 + 32 < 512) stage_ab(k0 + 32, buf ^ 1);

    v16bf a[2], b[2];
#pragma unroll
    for (int mt = 0; mt < 2; mt++) {
      const __bf16* row = &As[buf][mw * 32 + mt * 16 + l16][0];
      a[mt] = cat8(*(const v8bf*)(row + half * 8), *(const v8bf*)(row + 16 + half * 8));
    }
#pragma unroll
    for (int nt = 0; nt < 2; nt++) {
      const __bf16* row = &Bs[buf][nw * 32 + nt * 16 + l16][0];
      b[nt] = cat8(*(const v8bf*)(row + half * 8), *(const v8bf*)(row + 16 + half * 8));
    }
#pragma unroll
    for (int mt = 0; mt < 2; mt++)
#pragma unroll
      for (int nt = 0; nt < 2; nt++) acc[mt][nt] = WMMA_BF16(a[mt], b[nt], acc[mt][nt]);
  }

#pragma unroll
  for (int mt = 0; mt < 2; mt++)
#pragma unroll
    for (int nt = 0; nt < 2; nt++) {
      const int c = n_base + nw * 32 + nt * 16 + l16;
      const float bv_ = bo[c];
#pragma unroll
      for (int r = 0; r < 8; r++) {
        const int m = m_base + mw * 32 + mt * 16 + r + half * 8;
        out[(size_t)m * 512 + c] = acc[mt][nt][r] + bv_;
      }
    }
}

// ---------------------------------------------------------------------------
extern "C" void kernel_launch(void* const* d_in, const int* in_sizes, int n_in,
                              void* d_out, int out_size, void* d_ws, size_t ws_size,
                              hipStream_t stream) {
  (void)in_sizes; (void)n_in; (void)out_size; (void)ws_size;
  const float* q    = (const float*)d_in[0];
  const float* k    = (const float*)d_in[1];
  const float* v    = (const float*)d_in[2];
  const float* wq_w = (const float*)d_in[3];
  const float* wq_b = (const float*)d_in[4];
  const float* wk_w = (const float*)d_in[5];
  const float* wk_b = (const float*)d_in[6];
  const float* wv_w = (const float*)d_in[7];
  const float* wv_b = (const float*)d_in[8];
  const float* wo_w = (const float*)d_in[9];
  const float* wo_b = (const float*)d_in[10];

  const size_t N1 = (size_t)2 * 8 * 4096 * 64;  // head-split tensor elements
  const size_t NW = (size_t)512 * 512;          // weight matrix elements
  __bf16* qhw = (__bf16*)d_ws;
  __bf16* khw = qhw + N1;
  __bf16* vtw = khw + N1;
  __bf16* aow = vtw + N1;
  __bf16* wqb = aow + N1;
  __bf16* wkb = wqb + NW;
  __bf16* wvb = wkb + NW;
  __bf16* wob = wvb + NW;

  dim3 g0(128, 4);
  cvt_weights_kernel<<<g0, 256, 0, stream>>>(wq_w, wk_w, wv_w, wo_w, wqb, wkb, wvb, wob);
  dim3 g1(64, 8, 3);
  qkv_gemm_kernel<<<g1, 256, 0, stream>>>(q, k, v, wqb, wkb, wvb,
                                          wq_b, wk_b, wv_b, qhw, khw, vtw);
  dim3 g2(32, 16);
  attn_kernel<<<g2, 256, 0, stream>>>(qhw, khw, vtw, aow);
  dim3 g3(64, 8);
  out_proj_kernel<<<g3, 256, 0, stream>>>(aow, wob, wo_b, (float*)d_out);
}
